// SO2EquivariantGraphAttention_42004780155112
// MI455X (gfx1250) — matrix operand order repaired and checked by reading, not compiled, not run on hardware
//
#include <hip/hip_runtime.h>

typedef __attribute__((ext_vector_type(16))) __bf16 v16bf;
typedef __attribute__((ext_vector_type(8)))  float  v8f;

typedef unsigned short ushort_t;
typedef unsigned int   uint_t;

// ---------- small helpers ----------
__device__ __forceinline__ ushort_t f2bf(float f) {
  uint_t u = __float_as_uint(f);
  u += 0x7FFFu + ((u >> 16) & 1u);   // round-to-nearest-even
  return (ushort_t)(u >> 16);
}
__device__ __forceinline__ float bf2f(ushort_t h) {
  return __uint_as_float(((uint_t)h) << 16);
}
__device__ __forceinline__ float sigmoidf_(float x) { return 1.0f / (1.0f + __expf(-x)); }
__device__ __forceinline__ float siluf_(float x)    { return x * sigmoidf_(x); }
__device__ __forceinline__ float slrelu_(float x)   { return 0.6f * x + 0.4f * x * (2.0f * sigmoidf_(x) - 1.0f); }

__device__ __forceinline__ uint_t fkey(float f) {
  uint_t b = __float_as_uint(f);
  return (b & 0x80000000u) ? ~b : (b | 0x80000000u);
}
__device__ __forceinline__ float funkey(uint_t k) {
  uint_t b = (k & 0x80000000u) ? (k & 0x7FFFFFFFu) : ~k;
  return __uint_as_float(b);
}

// ---------- generic bf16 WMMA GEMM: C(MxN) = A(MxK) * B(KxN) [+bias] ----------
// Block tile 128x128, K-step 32, 256 threads = 8 waves in a 4x2 grid of 32x64
// wave tiles. Per wave per K-step: 2 A-frags x 4 B-frags -> 8 v_wmma, so the
// wmma : ds_load ratio is 8 : 12xb128 (4x denser than a 16x32 wave tile).
// Requires M % 128 == 0 and K % 32 == 0; N handled with zero-fill + guards.
union FragBF { uint4 u[2]; v16bf v; };

__global__ __launch_bounds__(256)
void gemm_bf16_kernel(const ushort_t* __restrict__ A, const ushort_t* __restrict__ B,
                      void* __restrict__ Cv, const float* __restrict__ bias,
                      int M, int N, int K, int lda, int ldb, int ldc, int outBf16)
{
  __shared__ ushort_t As[128][40];   // padded stride (80B) keeps 16B alignment, dodges conflicts
  __shared__ ushort_t Bs[128][40];   // B stored transposed: [n][k]

  const int t    = threadIdx.x;
  const int lane = t & 31;
  const int wave = t >> 5;
  const int wm   = wave >> 1;        // 0..3 -> rows wm*32..wm*32+31
  const int wn   = wave & 1;         // 0..1 -> cols wn*64..wn*64+63
  const int m0   = blockIdx.y << 7;
  const int n0   = blockIdx.x << 7;

  v8f acc[2][4] = {};

  const int lane15 = lane & 15;
  const int kb  = (lane < 16) ? 0 : 8;    // A frag K-base (ISA 16-bit A 16x32 layout)
  const int kb2 = (lane < 16) ? 0 : 8;    // B frag K-base (same K order as A frag)

  for (int k0 = 0; k0 < K; k0 += 32) {
    // A tile 128x32: 512 uint4 chunks, 2 per thread
#pragma unroll
    for (int i = 0; i < 2; ++i) {
      const int id = t * 2 + i;
      const int ar = id >> 2;
      const int ac = (id & 3) << 3;
      uint4 av = *(const uint4*)(A + (size_t)(m0 + ar) * lda + (k0 + ac));
      *(uint4*)&As[ar][ac] = av;
    }
    // B tile 32x128 -> transposed into Bs[n][k]; zero-fill beyond N
#pragma unroll
    for (int i = 0; i < 2; ++i) {
      const int id = t * 2 + i;
      const int bk = id >> 4;
      const int bn = (id & 15) << 3;
      uint4 bv = make_uint4(0u, 0u, 0u, 0u);
      if (n0 + bn < N)
        bv = *(const uint4*)(B + (size_t)(k0 + bk) * ldb + (n0 + bn));
      const ushort_t* bs = (const ushort_t*)&bv;
#pragma unroll
      for (int j = 0; j < 8; ++j) Bs[bn + j][bk] = bs[j];
    }
    __syncthreads();

    FragBF fa[2];
#pragma unroll
    for (int ai = 0; ai < 2; ++ai) {
      const int row = (wm << 5) + (ai << 4) + lane15;
      fa[ai].u[0] = *(const uint4*)&As[row][kb];        // K = kb..kb+7
      fa[ai].u[1] = *(const uint4*)&As[row][kb + 16];   // K = kb+16..kb+23
    }
#pragma unroll
    for (int bi = 0; bi < 4; ++bi) {
      const int col = (wn << 6) + (bi << 4) + lane15;
      FragBF fb;
      fb.u[0] = *(const uint4*)&Bs[col][kb2];
      fb.u[1] = *(const uint4*)&Bs[col][kb2 + 16];
#pragma unroll
      for (int ai = 0; ai < 2; ++ai)
        acc[ai][bi] = __builtin_amdgcn_wmma_f32_16x16x32_bf16(false, fa[ai].v, false, fb.v,
                                                              (short)0, acc[ai][bi], false, false);
    }
    __syncthreads();
  }

  const int rhalf = (lane < 16) ? 0 : 8;
#pragma unroll
  for (int ai = 0; ai < 2; ++ai) {
    const int rbase = m0 + (wm << 5) + (ai << 4) + rhalf;
#pragma unroll
    for (int bi = 0; bi < 4; ++bi) {
      const int col = n0 + (wn << 6) + (bi << 4) + lane15;
      if (col >= N) continue;
      const float bb = bias ? bias[col] : 0.0f;
#pragma unroll
      for (int i = 0; i < 8; ++i) {
        const int rr = rbase + i;
        float v = acc[ai][bi][i] + bb;
        if (outBf16) ((ushort_t*)Cv)[(size_t)rr * ldc + col] = f2bf(v);
        else         ((float*)Cv)[(size_t)rr * ldc + col] = v;
      }
    }
  }
}

// ---------- weight prep ----------
__global__ void convert_f32_bf16_kernel(const float* __restrict__ src, ushort_t* __restrict__ dst, int n) {
  int i = blockIdx.x * blockDim.x + threadIdx.x;
  if (i < n) dst[i] = f2bf(src[i]);
}

// Build [[wa,wb],[-wb,wa]] from src = [wa|wb] (rh x 2*ch) -> dst (2*rh x 2*ch)
__global__ void stack_complex_kernel(const float* __restrict__ src, ushort_t* __restrict__ dst,
                                     int rh, int ch) {
  int i = blockIdx.x * blockDim.x + threadIdx.x;
  int total = 4 * rh * ch;
  if (i >= total) return;
  int cols = 2 * ch;
  int r = i / cols, c = i - r * cols;
  float v;
  if (r < rh) v = src[(size_t)r * cols + c];
  else {
    int r2 = r - rh;
    v = (c < ch) ? -src[(size_t)r2 * cols + ch + c] : src[(size_t)r2 * cols + (c - ch)];
  }
  dst[i] = f2bf(v);
}

__global__ void zero_f32_kernel(float* p, int n) { int i = blockIdx.x * blockDim.x + threadIdx.x; if (i < n) p[i] = 0.0f; }
__global__ void zero_u32_kernel(uint_t* p, int n){ int i = blockIdx.x * blockDim.x + threadIdx.x; if (i < n) p[i] = 0u; }

// ---------- per-edge radial MLP layers 1+2 (192->64->64 with LN+silu) ----------
__device__ __forceinline__ float blk64_sum(float v, float* red, int j) {
  red[j] = v; __syncthreads();
#pragma unroll
  for (int s = 32; s > 0; s >>= 1) { if (j < s) red[j] += red[j + s]; __syncthreads(); }
  float r = red[0]; __syncthreads();
  return r;
}

__global__ __launch_bounds__(64)
void radial12_kernel(const float* __restrict__ edge_dist, const int* __restrict__ species,
                     const int* __restrict__ senders, const int* __restrict__ receivers,
                     const float* __restrict__ semb, const float* __restrict__ remb,
                     const float* __restrict__ w1, const float* __restrict__ b1,
                     const float* __restrict__ g1, const float* __restrict__ be1,
                     const float* __restrict__ w2, const float* __restrict__ b2,
                     const float* __restrict__ g2, const float* __restrict__ be2,
                     ushort_t* __restrict__ h2out)
{
  const int e = blockIdx.x, j = threadIdx.x;
  __shared__ float ee[192];
  __shared__ float h1s[64];
  __shared__ float red[64];
  const int ss = species[senders[e]];
  const int rs = species[receivers[e]];
  ee[j]        = edge_dist[(size_t)e * 64 + j];
  ee[64 + j]   = semb[ss * 64 + j];
  ee[128 + j]  = remb[rs * 64 + j];
  __syncthreads();

  float acc = b1[j];
  for (int i = 0; i < 192; ++i) acc += ee[i] * w1[i * 64 + j];
  float mu  = blk64_sum(acc, red, j) * (1.0f / 64.0f);
  float var = blk64_sum(acc * acc, red, j) * (1.0f / 64.0f) - mu * mu;
  float h1  = siluf_((acc - mu) * rsqrtf(var + 1e-5f) * g1[j] + be1[j]);
  h1s[j] = h1; __syncthreads();

  float acc2 = b2[j];
#pragma unroll
  for (int i = 0; i < 64; ++i) acc2 += h1s[i] * w2[i * 64 + j];
  mu  = blk64_sum(acc2, red, j) * (1.0f / 64.0f);
  var = blk64_sum(acc2 * acc2, red, j) * (1.0f / 64.0f) - mu * mu;
  float h2 = siluf_((acc2 - mu) * rsqrtf(var + 1e-5f) * g2[j] + be2[j]);
  h2out[(size_t)e * 64 + j] = f2bf(h2);
}

// ---------- Wigner rotate + x_edge scale -> packed bf16 SO2-conv-1 inputs ----------
__global__ __launch_bounds__(128)
void wigner_scale_kernel(const float* __restrict__ node_feats, const float* __restrict__ wigner,
                         const int* __restrict__ senders, const int* __restrict__ receivers,
                         const ushort_t* __restrict__ xe,
                         ushort_t* __restrict__ xs0, ushort_t* __restrict__ x1, ushort_t* __restrict__ x2)
{
  const int e = blockIdx.x, c = threadIdx.x;
  __shared__ float wig[475];   // 19 x 25
  for (int i = c; i < 475; i += 128) wig[i] = wigner[(size_t)e * 475 + i];
  const int snd = senders[e], rcv = receivers[e];
  const float* base = (c < 64) ? (node_feats + (size_t)snd * 1600 + c)
                               : (node_feats + (size_t)rcv * 1600 + (c - 64));
  float f[25];
#pragma unroll
  for (int k = 0; k < 25; ++k) f[k] = base[k * 64];
  __syncthreads();

  float msg[19];
#pragma unroll
  for (int m = 0; m < 19; ++m) {
    float a = 0.0f;
#pragma unroll
    for (int k = 0; k < 25; ++k) a += wig[m * 25 + k] * f[k];
    msg[m] = a;
  }
  const ushort_t* xee = xe + (size_t)e * 1536;
#pragma unroll
  for (int l = 0; l < 5; ++l)
    xs0[(size_t)e * 640 + l * 128 + c] = f2bf(msg[l] * bf2f(xee[l * 128 + c]));
#pragma unroll
  for (int j = 0; j < 4; ++j) {
    float s = bf2f(xee[640 + j * 128 + c]);
    x1[(size_t)e * 1024 + j * 128 + c]       = f2bf(msg[5 + j] * s);
    x1[(size_t)e * 1024 + 512 + j * 128 + c] = f2bf(msg[9 + j] * s);
  }
#pragma unroll
  for (int j = 0; j < 3; ++j) {
    float s = bf2f(xee[1152 + j * 128 + c]);
    x2[(size_t)e * 768 + j * 128 + c]       = f2bf(msg[13 + j] * s);
    x2[(size_t)e * 768 + 384 + j * 128 + c] = f2bf(msg[16 + j] * s);
  }
}

// ---------- grid silu round-trip + gate -> packed bf16 SO2-conv-2 inputs ----------
__global__ __launch_bounds__(64)
void grid_gate_kernel(const ushort_t* __restrict__ y0, const ushort_t* __restrict__ y1,
                      const ushort_t* __restrict__ y2,
                      const float* __restrict__ to_grid, const float* __restrict__ from_grid,
                      ushort_t* __restrict__ g0, ushort_t* __restrict__ g1, ushort_t* __restrict__ g2)
{
  const int e = blockIdx.x, c = threadIdx.x;
  __shared__ float tg[1900];  // 100x19
  __shared__ float fg[1900];  // 19x100
  for (int i = c; i < 1900; i += 64) { tg[i] = to_grid[i]; fg[i] = from_grid[i]; }
  float m1v[19];
#pragma unroll
  for (int l = 0; l < 5; ++l)   m1v[l] = bf2f(y0[(size_t)e * 640 + l * 64 + c]);
#pragma unroll
  for (int m = 5; m < 13; ++m)  m1v[m] = bf2f(y1[(size_t)e * 512 + (m - 5) * 64 + c]);
#pragma unroll
  for (int m = 13; m < 19; ++m) m1v[m] = bf2f(y2[(size_t)e * 384 + (m - 13) * 64 + c]);
  __syncthreads();

  float tens[19];
#pragma unroll
  for (int m = 0; m < 19; ++m) tens[m] = 0.0f;
  for (int p = 0; p < 100; ++p) {
    float g = 0.0f;
#pragma unroll
    for (int m = 0; m < 19; ++m) g += tg[p * 19 + m] * m1v[m];
    g = siluf_(g);
#pragma unroll
    for (int m = 0; m < 19; ++m) tens[m] += fg[m * 100 + p] * g;
  }
  float gate = bf2f(y0[(size_t)e * 640 + 576 + c]);
  g0[(size_t)e * 320 + c] = f2bf(siluf_(gate));
#pragma unroll
  for (int l = 1; l < 5; ++l)   g0[(size_t)e * 320 + l * 64 + c]        = f2bf(tens[l]);
#pragma unroll
  for (int m = 5; m < 13; ++m)  g1[(size_t)e * 512 + (m - 5) * 64 + c]  = f2bf(tens[m]);
#pragma unroll
  for (int m = 13; m < 19; ++m) g2[(size_t)e * 384 + (m - 13) * 64 + c] = f2bf(tens[m]);
}

// ---------- alpha: per-(edge,head) LN(32) + smooth-leaky-relu + dot ----------
__global__ __launch_bounds__(256)
void alpha_logits_kernel(const ushort_t* __restrict__ y0,
                         const float* __restrict__ aln_g, const float* __restrict__ aln_b,
                         const float* __restrict__ alpha_dot, float* __restrict__ logits)
{
  const int pair = blockIdx.x * 8 + (threadIdx.x >> 5);
  const int lane = threadIdx.x & 31;
  const int e = pair >> 3, h = pair & 7;
  float v = bf2f(y0[(size_t)e * 640 + 320 + h * 32 + lane]);
  float s = v;
#pragma unroll
  for (int m = 16; m > 0; m >>= 1) s += __shfl_xor(s, m, 32);
  float mu = s * (1.0f / 32.0f);
  float d = v - mu;
  float q = d * d;
#pragma unroll
  for (int m = 16; m > 0; m >>= 1) q += __shfl_xor(q, m, 32);
  float var = q * (1.0f / 32.0f);
  float a = slrelu_(d * rsqrtf(var + 1e-5f) * aln_g[lane] + aln_b[lane]);
  float p = a * alpha_dot[h * 32 + lane];
#pragma unroll
  for (int m = 16; m > 0; m >>= 1) p += __shfl_xor(p, m, 32);
  if (lane == 0) logits[(size_t)e * 8 + h] = p;
}

__global__ void seg_max_kernel(const float* __restrict__ logits, const int* __restrict__ receivers,
                               uint_t* __restrict__ nmax, int total) {
  int i = blockIdx.x * blockDim.x + threadIdx.x;
  if (i >= total) return;
  int e = i >> 3, h = i & 7;
  atomicMax(&nmax[receivers[e] * 8 + h], fkey(logits[i]));
}
__global__ void seg_exp_kernel(float* __restrict__ logits, const int* __restrict__ receivers,
                               const uint_t* __restrict__ nmax, float* __restrict__ nsum, int total) {
  int i = blockIdx.x * blockDim.x + threadIdx.x;
  if (i >= total) return;
  int e = i >> 3, h = i & 7;
  float m = funkey(nmax[receivers[e] * 8 + h]);
  if (!(m > -1e30f && m < 1e30f)) m = 0.0f;
  float ex = __expf(logits[i] - m);
  logits[i] = ex;
  atomicAdd(&nsum[receivers[e] * 8 + h], ex);
}
__global__ void seg_norm_kernel(const float* __restrict__ logits, const int* __restrict__ receivers,
                                const float* __restrict__ nsum, float* __restrict__ alpha, int total) {
  int i = blockIdx.x * blockDim.x + threadIdx.x;
  if (i >= total) return;
  int e = i >> 3, h = i & 7;
  alpha[i] = logits[i] / (nsum[receivers[e] * 8 + h] + 1e-16f);
}

// ---------- alpha-scale + Wigner^T + scatter-add into node accumulator ----------
__global__ __launch_bounds__(128)
void attn_scatter_kernel(const ushort_t* __restrict__ z0, const ushort_t* __restrict__ z1,
                         const ushort_t* __restrict__ z2, const float* __restrict__ alpha,
                         const float* __restrict__ wigner, const int* __restrict__ receivers,
                         float* __restrict__ node)
{
  const int e = blockIdx.x, c = threadIdx.x;
  __shared__ float wig[475];
  for (int i = c; i < 475; i += 128) wig[i] = wigner[(size_t)e * 475 + i];
  const float al = alpha[(size_t)e * 8 + (c >> 4)];
  float m3[19];
#pragma unroll
  for (int m = 0; m < 5; ++m)   m3[m] = bf2f(z0[(size_t)e * 640 + m * 128 + c]) * al;
#pragma unroll
  for (int m = 5; m < 13; ++m)  m3[m] = bf2f(z1[(size_t)e * 1024 + (m - 5) * 128 + c]) * al;
#pragma unroll
  for (int m = 13; m < 19; ++m) m3[m] = bf2f(z2[(size_t)e * 768 + (m - 13) * 128 + c]) * al;
  __syncthreads();
  float* dst = node + (size_t)receivers[e] * 3200 + c;
#pragma unroll
  for (int k = 0; k < 25; ++k) {
    float a = 0.0f;
#pragma unroll
    for (int m = 0; m < 19; ++m) a += wig[m * 25 + k] * m3[m];
    atomicAdd(dst + k * 128, a);
  }
}

// ---------- driver ----------
extern "C" void kernel_launch(void* const* d_in, const int* in_sizes, int n_in,
                              void* d_out, int out_size, void* d_ws, size_t ws_size,
                              hipStream_t stream)
{
  (void)in_sizes; (void)n_in; (void)out_size; (void)ws_size;
  const float* node_feats = (const float*)d_in[0];
  const int*   species    = (const int*)d_in[1];
  const float* edge_dist  = (const float*)d_in[2];
  const int*   senders    = (const int*)d_in[3];
  const int*   receivers  = (const int*)d_in[4];
  const float* wigner     = (const float*)d_in[5];
  const float* snd_emb    = (const float*)d_in[6];
  const float* rcv_emb    = (const float*)d_in[7];
  const float* rad_w1  = (const float*)d_in[8];
  const float* rad_b1  = (const float*)d_in[9];
  const float* rad_g1  = (const float*)d_in[10];
  const float* rad_be1 = (const float*)d_in[11];
  const float* rad_w2  = (const float*)d_in[12];
  const float* rad_b2  = (const float*)d_in[13];
  const float* rad_g2  = (const float*)d_in[14];
  const float* rad_be2 = (const float*)d_in[15];
  const float* rad_w3  = (const float*)d_in[16];
  const float* rad_b3  = (const float*)d_in[17];
  const float* c1_w0   = (const float*)d_in[18];
  const float* c1_b0   = (const float*)d_in[19];
  const float* c1_wm1  = (const float*)d_in[20];
  const float* c1_wm2  = (const float*)d_in[21];
  const float* aln_g   = (const float*)d_in[22];
  const float* aln_b   = (const float*)d_in[23];
  const float* alpha_d = (const float*)d_in[24];
  const float* to_grid = (const float*)d_in[25];
  const float* from_gr = (const float*)d_in[26];
  const float* c2_w0   = (const float*)d_in[27];
  const float* c2_b0   = (const float*)d_in[28];
  const float* c2_wm1  = (const float*)d_in[29];
  const float* c2_wm2  = (const float*)d_in[30];
  const float* proj_w  = (const float*)d_in[31];
  const float* proj_b  = (const float*)d_in[32];
  float* out = (float*)d_out;

  const int E = 24576, NN = 4096;
  char* ws = (char*)d_ws;
  size_t off = 0;
  auto alloc = [&](size_t bytes) -> char* {
    char* p = ws + off;
    off += (bytes + 255) & ~(size_t)255;
    return p;
  };
  auto mx = [](size_t a, size_t b) { return a > b ? a : b; };

  ushort_t* wb_c1w0 = (ushort_t*)alloc((size_t)640 * 640 * 2);
  ushort_t* wb_c1m1 = (ushort_t*)alloc((size_t)1024 * 512 * 2);
  ushort_t* wb_c1m2 = (ushort_t*)alloc((size_t)768 * 384 * 2);
  ushort_t* wb_c2w0 = (ushort_t*)alloc((size_t)320 * 640 * 2);
  ushort_t* wb_c2m1 = (ushort_t*)alloc((size_t)512 * 1024 * 2);
  ushort_t* wb_c2m2 = (ushort_t*)alloc((size_t)384 * 768 * 2);
  ushort_t* wb_rad3 = (ushort_t*)alloc((size_t)64 * 1536 * 2);
  ushort_t* wb_proj = (ushort_t*)alloc((size_t)5 * 128 * 64 * 2);
  ushort_t* h2bf    = (ushort_t*)alloc((size_t)E * 64 * 2);

  char* rXE = alloc(mx((size_t)E * 1536 * 2, (size_t)NN * 3200 * 4)); // x_edge then node f32
  ushort_t* xe_bf  = (ushort_t*)rXE;
  float*    node_f = (float*)rXE;

  ushort_t* xs0 = (ushort_t*)alloc((size_t)E * 640 * 2);   // later z0
  ushort_t* x1  = (ushort_t*)alloc((size_t)E * 1024 * 2);  // later z1
  ushort_t* x2  = (ushort_t*)alloc((size_t)E * 768 * 2);   // later z2
  ushort_t *z0 = xs0, *z1 = x1, *z2 = x2;

  char* rY0 = alloc(mx((size_t)E * 640 * 2, (size_t)NN * 3200 * 2)); // y0ext then node bf16
  ushort_t* y0      = (ushort_t*)rY0;
  ushort_t* node_bf = (ushort_t*)rY0;
  ushort_t* y1 = (ushort_t*)alloc((size_t)E * 512 * 2);
  ushort_t* y2 = (ushort_t*)alloc((size_t)E * 384 * 2);
  ushort_t* g0 = (ushort_t*)alloc((size_t)E * 320 * 2);
  ushort_t* g1 = (ushort_t*)alloc((size_t)E * 512 * 2);
  ushort_t* g2 = (ushort_t*)alloc((size_t)E * 384 * 2);
  float*  logits = (float*)alloc((size_t)E * 8 * 4);
  float*  alpha  = (float*)alloc((size_t)E * 8 * 4);
  uint_t* nmax   = (uint_t*)alloc((size_t)NN * 8 * 4);
  float*  nsum   = (float*)alloc((size_t)NN * 8 * 4);

  auto cdiv = [](int a, int b) { return (a + b - 1) / b; };

  // ---- weight prep (f32 -> bf16, complex stacking) ----
  convert_f32_bf16_kernel<<<cdiv(640 * 640, 256), 256, 0, stream>>>(c1_w0, wb_c1w0, 640 * 640);
  convert_f32_bf16_kernel<<<cdiv(320 * 640, 256), 256, 0, stream>>>(c2_w0, wb_c2w0, 320 * 640);
  convert_f32_bf16_kernel<<<cdiv(64 * 1536, 256), 256, 0, stream>>>(rad_w3, wb_rad3, 64 * 1536);
  convert_f32_bf16_kernel<<<cdiv(5 * 128 * 64, 256), 256, 0, stream>>>(proj_w, wb_proj, 5 * 128 * 64);
  stack_complex_kernel<<<cdiv(1024 * 512, 256), 256, 0, stream>>>(c1_wm1, wb_c1m1, 512, 256);
  stack_complex_kernel<<<cdiv(768 * 384, 256), 256, 0, stream>>>(c1_wm2, wb_c1m2, 384, 192);
  stack_complex_kernel<<<cdiv(512 * 1024, 256), 256, 0, stream>>>(c2_wm1, wb_c2m1, 256, 512);
  stack_complex_kernel<<<cdiv(384 * 768, 256), 256, 0, stream>>>(c2_wm2, wb_c2m2, 192, 384);

  // ---- radial MLP layers 1+2, then layer 3 as WMMA GEMM (E x 1536) ----
  radial12_kernel<<<E, 64, 0, stream>>>(edge_dist, species, senders, receivers, snd_emb, rcv_emb,
                                        rad_w1, rad_b1, rad_g1, rad_be1,
                                        rad_w2, rad_b2, rad_g2, rad_be2, h2bf);
  gemm_bf16_kernel<<<dim3(cdiv(1536, 128), E / 128), 256, 0, stream>>>(
      h2bf, wb_rad3, xe_bf, rad_b3, E, 1536, 64, 64, 1536, 1536, 1);

  // ---- Wigner rotate + scale -> SO2 conv 1 inputs, then 3 WMMA GEMMs ----
  wigner_scale_kernel<<<E, 128, 0, stream>>>(node_feats, wigner, senders, receivers, xe_bf, xs0, x1, x2);
  gemm_bf16_kernel<<<dim3(cdiv(640, 128), E / 128), 256, 0, stream>>>(
      xs0, wb_c1w0, y0, c1_b0, E, 640, 640, 640, 640, 640, 1);
  gemm_bf16_kernel<<<dim3(cdiv(512, 128), E / 128), 256, 0, stream>>>(
      x1, wb_c1m1, y1, nullptr, E, 512, 1024, 1024, 512, 512, 1);
  gemm_bf16_kernel<<<dim3(cdiv(384, 128), E / 128), 256, 0, stream>>>(
      x2, wb_c1m2, y2, nullptr, E, 384, 768, 768, 384, 384, 1);

  // ---- grid round-trip + gate, attention logits + segment softmax ----
  grid_gate_kernel<<<E, 64, 0, stream>>>(y0, y1, y2, to_grid, from_gr, g0, g1, g2);
  alpha_logits_kernel<<<E, 256, 0, stream>>>(y0, aln_g, aln_b, alpha_d, logits);
  zero_u32_kernel<<<cdiv(NN * 8, 256), 256, 0, stream>>>(nmax, NN * 8);
  zero_f32_kernel<<<cdiv(NN * 8, 256), 256, 0, stream>>>(nsum, NN * 8);
  seg_max_kernel<<<cdiv(E * 8, 256), 256, 0, stream>>>(logits, receivers, nmax, E * 8);
  seg_exp_kernel<<<cdiv(E * 8, 256), 256, 0, stream>>>(logits, receivers, nmax, nsum, E * 8);
  seg_norm_kernel<<<cdiv(E * 8, 256), 256, 0, stream>>>(logits, receivers, nsum, alpha, E * 8);

  // ---- SO2 conv 2 as 3 WMMA GEMMs (z* aliases x* storage) ----
  gemm_bf16_kernel<<<dim3(cdiv(640, 128), E / 128), 256, 0, stream>>>(
      g0, wb_c2w0, z0, c2_b0, E, 640, 320, 320, 640, 640, 1);
  gemm_bf16_kernel<<<dim3(cdiv(1024, 128), E / 128), 256, 0, stream>>>(
      g1, wb_c2m1, z1, nullptr, E, 1024, 512, 512, 1024, 1024, 1);
  gemm_bf16_kernel<<<dim3(cdiv(768, 128), E / 128), 256, 0, stream>>>(
      g2, wb_c2m2, z2, nullptr, E, 768, 384, 384, 768, 768, 1);

  // ---- alpha-scale + Wigner^T + scatter into node accumulator ----
  zero_f32_kernel<<<cdiv(NN * 3200, 256), 256, 0, stream>>>(node_f, NN * 3200);
  attn_scatter_kernel<<<E, 128, 0, stream>>>(z0, z1, z2, alpha, wigner, receivers, node_f);
  convert_f32_bf16_kernel<<<cdiv(NN * 3200, 256), 256, 0, stream>>>(node_f, node_bf, NN * 3200);

  // ---- per-degree output projection: 25 small WMMA GEMMs into d_out ----
  static const int l_of_k[25] = {0, 1,1,1, 2,2,2,2,2, 3,3,3,3,3,3,3, 4,4,4,4,4,4,4,4,4};
  for (int k = 0; k < 25; ++k) {
    gemm_bf16_kernel<<<dim3(1, NN / 128), 256, 0, stream>>>(
        node_bf + k * 128, wb_proj + (size_t)l_of_k[k] * 128 * 64,
        out + k * 64, (k == 0) ? proj_b : nullptr,
        NN, 64, 128, 3200, 64, 1600, 0);
  }
}
